// Dynamics_43911745634481
// MI455X (gfx1250) — matrix operand order repaired and checked by reading, not compiled, not run on hardware
//
#include <hip/hip_runtime.h>

// ============================================================================
// MI455X (gfx1250) batched barotropic-vorticity solver.
// Per step: K12 (CNN forcing via f16 WMMA  ||  row FFT fwd -> transposed Zt)
//           K3  (col FFT fwd + spectral + 2 packed col inverse FFTs, tiled I/O)
//           K4  (row inverse FFTs + pointwise update + optional snapshot)
// All global accesses coalesced; intermediates (~16MB) live in d_ws (L2-hot).
// ============================================================================

typedef __attribute__((ext_vector_type(16))) _Float16 v16h;
typedef __attribute__((ext_vector_type(2)))  _Float16 v2h;
typedef __attribute__((ext_vector_type(8)))  float    v8f;

#define GRIDN   256
#define NPIX    65536
#define BATCH   8
#define DT      0.01f
#define R_DRAG  0.05f
#define BETA    0.5f
#define Q_SCALE 0.01f

__device__ __forceinline__ v8f wmma_f16(v16h a, v16h b, v8f c) {
  // D = A(16x32 f16) x B(32x16 f16) + C(16x16 f32)
  return __builtin_amdgcn_wmma_f32_16x16x32_f16(false, a, false, b, (short)0, c,
                                                false, false);
}

// ---------------------------------------------------------------------------
// 256-point FFT, four-step 16x16, one transform per 256-thread block.
// n = 16*n1 + n2, k = k1 + 16*k2:
//   X[k1+16k2] = sum_n2 W256^{n2 k1} W16^{n2 k2} ( sum_n1 x[16n1+n2] W16^{n1 k1} )
// tw[m] = e^{-2 pi i m/256}; inverse uses the conjugate. 1/N^2 folded into K3.
// Caller must __syncthreads() between consecutive calls (sr/si reuse).
// ---------------------------------------------------------------------------
__device__ __forceinline__ void fft256(float xr, float xi,
                                       float* __restrict__ sr, float* __restrict__ si,
                                       const float* __restrict__ twr,
                                       const float* __restrict__ twi,
                                       int t, bool inverse,
                                       float* outr, float* outi)
{
  sr[t] = xr; si[t] = xi;
  __syncthreads();
  const int k1 = t & 15, n2 = t >> 4;
  float ar = 0.f, ai = 0.f;
#pragma unroll
  for (int n1 = 0; n1 < 16; ++n1) {
    int m = ((n1 * k1) & 15) << 4;                 // W16^{n1 k1}
    float c = twr[m], s = inverse ? -twi[m] : twi[m];
    float vr = sr[(n1 << 4) + n2], vi = si[(n1 << 4) + n2];
    ar += vr * c - vi * s;
    ai += vr * s + vi * c;
  }
  { // twiddle W256^{n2 k1}
    int m = (n2 * k1) & 255;
    float c = twr[m], s = inverse ? -twi[m] : twi[m];
    float tr = ar * c - ai * s;
    ai = ar * s + ai * c;
    ar = tr;
  }
  __syncthreads();
  sr[(k1 << 4) + n2] = ar; si[(k1 << 4) + n2] = ai;
  __syncthreads();
  const int k2 = t >> 4;          // output index k = (t&15) + 16*(t>>4) = t
  float br = 0.f, bi = 0.f;
#pragma unroll
  for (int q = 0; q < 16; ++q) {
    int m = ((q * k2) & 15) << 4;                  // W16^{n2 k2}
    float c = twr[m], s = inverse ? -twi[m] : twi[m];
    float vr = sr[(k1 << 4) + q], vi = si[(k1 << 4) + q];
    br += vr * c - vi * s;
    bi += vr * s + vi * c;
  }
  *outr = br; *outi = bi;
}

__device__ __forceinline__ void tw_init(float* twr, float* twi, int t) {
  float a = -0.0245436926061702596f * (float)t;    // -2*pi/256 * t
  float s_, c_; sincosf(a, &s_, &c_);
  twr[t] = c_; twi[t] = s_;                        // published by fft256's 1st sync
}

// ---------------------------------------------------------------------------
// K0: state init  z = IN_W * zeta   (IN_W == 1)
// ---------------------------------------------------------------------------
__global__ void k0_init(const float* __restrict__ zeta, float* __restrict__ z) {
  int i = blockIdx.x * 256 + threadIdx.x;
  z[i] = zeta[i];
}

// ---------------------------------------------------------------------------
// K12: heterogeneous grid, 256 threads/block, gridDim.x = 272:
//   blockIdx.x < 256  -> CNN forcing tile (16x16 output pixels)
//   blockIdx.x >= 256 -> 16-row group of forward x-FFTs, stored TRANSPOSED:
//                        Zt[b][kx][y]  (coalesced via LDS tile transpose)
//
// CNN:  Q = Q0 + 0.01 * ( conv2( relu( conv1(z)+b1 ) ) + b2 )
// WMMA layout assumptions (gfx1250 wave32):
//   A (f16 16xK): lane<16: e0..15->K=e (0..15)
//                 lane>=16: e0..15->K=e+16 (16..31)
//   B (f16 Kx16): lane n<16: col n, K=0..15 at e=K; lane n+16: K=16..31 at e=K-16
//   C/D (f32):    VGPR r: lane<16 -> (M=r,N=lane); lane>=16 -> (M=r+8,N=lane-16)
// ---------------------------------------------------------------------------
__global__ __launch_bounds__(256) void k12_cnn_rows(const float* __restrict__ z,
                                                    const float* __restrict__ Q0,
                                                    const float* __restrict__ W1,
                                                    const float* __restrict__ b1,
                                                    const float* __restrict__ W2,
                                                    const float* __restrict__ b2,
                                                    float* __restrict__ Qout,
                                                    float2* __restrict__ Zt)
{
  __shared__ __align__(16) char smem[39680];
  const int tid = threadIdx.x;
  const int b = blockIdx.y;
  const int bx = blockIdx.x;

  if (bx < 256) {
    // =========================== CNN branch ===============================
    _Float16* zT = (_Float16*)smem;            // 20x20 z tile + halo2  (800 B)
    _Float16* hT = (_Float16*)(smem + 800);    // 18x18 region, ch stride 34 (22032 B)
    float*    dT = (float*)(smem + 22832);     // 18x18 region, j stride 13  (16848 B)

    const int ty0 = (bx >> 4) << 4, tx0 = (bx & 15) << 4;
    const int lane = tid & 31, wv = tid >> 5;
    const int lm = lane & 15;
    const bool hiHalf = lane >= 16;
    const float* zb = z + b * NPIX;

    // static A matrices + bias C (per-lane registers)
    v16h aW1lo, aW1hi, aW2;
#pragma unroll
    for (int e = 0; e < 16; ++e) {
      int K = e + (hiHalf ? 16 : 0);
      aW1lo[e] = (K < 9) ? (_Float16)W1[K * 32 + lm]      : (_Float16)0.f;
      aW1hi[e] = (K < 9) ? (_Float16)W1[K * 32 + lm + 16] : (_Float16)0.f;
      aW2[e]   = (lm < 9) ? (_Float16)W2[lm * 32 + K]     : (_Float16)0.f;
    }
    v8f cb1lo, cb1hi;
#pragma unroll
    for (int r = 0; r < 8; ++r) {
      int c0 = r + (hiHalf ? 8 : 0);
      cb1lo[r] = b1[c0];
      cb1hi[r] = b1[c0 + 16];
    }

    // phase 1: z tile (zero-pad outside image == conv1 SAME pad)
    for (int i = tid; i < 400; i += 256) {
      int zy = i / 20, zx = i % 20;
      int gy = ty0 + zy - 2, gx = tx0 + zx - 2;
      float v = (gy >= 0 && gy < GRIDN && gx >= 0 && gx < GRIDN) ? zb[gy * 256 + gx] : 0.f;
      zT[zy * 20 + zx] = (_Float16)v;
    }
    __syncthreads();

    // phase 2: conv1 + relu over 18x18 region (36 wave-groups over 8 waves)
    for (int g = wv; g < 36; g += 8) {
      int ry = g >> 1;
      int px = ((g & 1) << 4) + lm;
      bool pxok = px < 18;
      v16h B = {};
      if (!hiHalf && pxok) {
        int zy = ry + 1, zx = px + 1;
#pragma unroll
        for (int j = 0; j < 9; ++j) {
          int dy = j / 3 - 1, dx = j % 3 - 1;
          B[j] = zT[(zy + dy) * 20 + (zx + dx)];
        }
      }
      v8f dlo = wmma_f16(aW1lo, B, cb1lo);   // channels 0..15
      v8f dhi = wmma_f16(aW1hi, B, cb1hi);   // channels 16..31
      if (pxok) {
        int base = (ry * 18 + px) * 34 + (hiHalf ? 8 : 0);
#pragma unroll
        for (int r = 0; r < 8; r += 2) {
          v2h p; p[0] = (_Float16)fmaxf(dlo[r], 0.f); p[1] = (_Float16)fmaxf(dlo[r + 1], 0.f);
          *(v2h*)&hT[base + r] = p;
          v2h q; q[0] = (_Float16)fmaxf(dhi[r], 0.f); q[1] = (_Float16)fmaxf(dhi[r + 1], 0.f);
          *(v2h*)&hT[base + 16 + r] = q;
        }
      }
    }
    __syncthreads();

    // phase 3: Dmat[p][j] = sum_c h[p][c] * W2[j][c]  (one WMMA, K=32 exact)
    for (int g = wv; g < 36; g += 8) {
      int ry = g >> 1;
      int px = ((g & 1) << 4) + lm;
      bool pxok = px < 18;
      v16h B = {};
      if (pxok) {
        int base = (ry * 18 + px) * 34 + (hiHalf ? 16 : 0);
#pragma unroll
        for (int i = 0; i < 8; ++i) {
          v2h p = *(const v2h*)&hT[base + 2 * i];
          B[2 * i] = p[0]; B[2 * i + 1] = p[1];
        }
      }
      v8f zero = {};
      v8f d2 = wmma_f16(aW2, B, zero);
      if (pxok) {
        int dbase = (ry * 18 + px) * 13;
        if (!hiHalf) {
#pragma unroll
          for (int r = 0; r < 8; ++r) dT[dbase + r] = d2[r];   // j = 0..7
        } else {
          dT[dbase + 8] = d2[0];                                // j = 8
        }
      }
    }
    __syncthreads();

    // phase 4: tap gather (zero-pad of h outside image == conv2 SAME pad)
    {
      int ly = tid >> 4, lx = tid & 15;
      int gy = ty0 + ly, gx = tx0 + lx;
      float q = b2[0];
#pragma unroll
      for (int j = 0; j < 9; ++j) {
        int dy = j / 3 - 1, dx = j % 3 - 1;
        int sy = gy + dy, sx = gx + dx;
        if (sy >= 0 && sy < GRIDN && sx >= 0 && sx < GRIDN)
          q += dT[((ly + dy + 1) * 18 + (lx + dx + 1)) * 13 + j];
      }
      Qout[b * NPIX + gy * 256 + gx] = Q0[gy * 256 + gx] + Q_SCALE * q;
    }
  } else {
    // ===================== forward row-FFT branch =========================
    float* twr = (float*)smem;                 // 256
    float* twi = twr + 256;
    float* sr  = twi + 256;
    float* si  = sr + 256;
    float* tr_ = si + 256;                     // 16 x 257 tile (real)
    float* ti_ = tr_ + 16 * 257;               // 16 x 257 tile (imag)

    const int t = tid;
    const int y0 = (bx - 256) << 4;
    tw_init(twr, twi, t);

    for (int r = 0; r < 16; ++r) {
      float xin = z[(b * 256 + y0 + r) * 256 + t];     // coalesced row read
      if (r < 15)                                      // overlap next row fetch
        __builtin_prefetch(&z[(b * 256 + y0 + r + 1) * 256 + t], 0, 0);
      float or_, oi_;
      fft256(xin, 0.f, sr, si, twr, twi, t, false, &or_, &oi_);
      tr_[r * 257 + t] = or_;
      ti_[r * 257 + t] = oi_;
      __syncthreads();                                  // sr reuse + publish tile
    }
    // transposed coalesced store: Zt[b][kx][y]
    for (int c = 0; c < 16; ++c) {
      int kx = (c << 4) + (t >> 4), yl = t & 15;
      Zt[(b * 256 + kx) * 256 + (y0 + yl)] =
          make_float2(tr_[yl * 257 + kx], ti_[yl * 257 + kx]);
    }
  }
}

// ---------------------------------------------------------------------------
// K3: 8 columns per block. For each column kx: forward FFT_y of Zt[b][kx][:]
// (contiguous read), spectral factors, two packed inverse FFT_y transforms:
//   S1 = IFFTy((uh + i*vh)/N^2), S2 = IFFTy((zxh + i*zyh)/N^2)
// Results staged in padded LDS tiles, stored transposed -> S[b][y][kx]
// (contiguous for K4's row reads).
// ---------------------------------------------------------------------------
__global__ __launch_bounds__(256) void k3_cols(const float2* __restrict__ Zt,
                                               float2* __restrict__ S1,
                                               float2* __restrict__ S2)
{
  __shared__ float twr[256], twi[256], sr[256], si[256];
  __shared__ float t1r[8 * 257], t1i[8 * 257];
  __shared__ float t2r[8 * 257], t2i[8 * 257];

  const int t = threadIdx.x, b = blockIdx.y;
  const int kx0 = blockIdx.x << 3;
  tw_init(twr, twi, t);
  const float ky = (t < 128) ? (float)t : (float)(t - 256);
  const float sc = 1.f / 65536.f;                    // full 1/N^2 of ifft2

  for (int r = 0; r < 8; ++r) {
    const int kx = kx0 + r;
    float2 zin = Zt[(b * 256 + kx) * 256 + t];       // coalesced column read
    if (r < 7)                                       // overlap next column fetch
      __builtin_prefetch(&Zt[(b * 256 + kx + 1) * 256 + t], 0, 0);
    float Zr, Zi;
    fft256(zin.x, zin.y, sr, si, twr, twi, t, false, &Zr, &Zi);

    float kxf = (kx < 128) ? (float)kx : (float)(kx - 256);
    float K2v = kxf * kxf + ky * ky;
    float invlap = (K2v > 0.f) ? (-1.f / K2v) : 0.f;
    float phr = Zr * invlap, phi = Zi * invlap;
    // A1 = uh + i vh ; uh = -i*ky*ph ; vh = i*kx*ph
    float A1r = ( ky * phi - kxf * phr) * sc;
    float A1i = (-ky * phr - kxf * phi) * sc;
    // A2 = zxh + i zyh ; zxh = i*kx*Zh ; zyh = i*ky*Zh
    float A2r = (-kxf * Zi - ky * Zr) * sc;
    float A2i = ( kxf * Zr - ky * Zi) * sc;

    __syncthreads();
    float s1r, s1i;
    fft256(A1r, A1i, sr, si, twr, twi, t, true, &s1r, &s1i);
    t1r[r * 257 + t] = s1r; t1i[r * 257 + t] = s1i;
    __syncthreads();
    float s2r, s2i;
    fft256(A2r, A2i, sr, si, twr, twi, t, true, &s2r, &s2i);
    t2r[r * 257 + t] = s2r; t2i[r * 257 + t] = s2i;
    __syncthreads();                                  // sr reuse + publish tiles
  }
  // transposed coalesced stores: S[b][y][kx0+xl]
  for (int c = 0; c < 8; ++c) {
    int y = (c << 5) + (t >> 3), xl = t & 7;
    S1[(b * 256 + y) * 256 + kx0 + xl] = make_float2(t1r[xl * 257 + y], t1i[xl * 257 + y]);
    S2[(b * 256 + y) * 256 + kx0 + xl] = make_float2(t2r[xl * 257 + y], t2i[xl * 257 + y]);
  }
}

// ---------------------------------------------------------------------------
// K4: per row y: inverse FFT_x of S1 -> (u,v), of S2 -> (zx,zy); update z;
// optionally write the snapshot (OUT_W == 1).
// ---------------------------------------------------------------------------
__global__ __launch_bounds__(256) void k4_rows_inv_update(const float2* __restrict__ S1,
                                                          const float2* __restrict__ S2,
                                                          const float* __restrict__ Q,
                                                          float* __restrict__ z,
                                                          float* __restrict__ out,
                                                          int oidx)
{
  __shared__ float twr[256], twi[256], sr[256], si[256];
  const int t = threadIdx.x, y = blockIdx.x, b = blockIdx.y;
  tw_init(twr, twi, t);

  const int idx = (b * 256 + y) * 256 + t;
  // warm caches for operands of the second FFT / pointwise phase
  __builtin_prefetch(&S2[idx], 0, 0);
  __builtin_prefetch(&Q[idx], 0, 0);
  __builtin_prefetch(&z[idx], 0, 0);

  float2 a = S1[idx];
  float u, v;
  fft256(a.x, a.y, sr, si, twr, twi, t, true, &u, &v);
  __syncthreads();
  float2 c = S2[idx];
  float zx, zy;
  fft256(c.x, c.y, sr, si, twr, twi, t, true, &zx, &zy);

  float zo = z[idx];
  float adv = u * zx + v * zy;
  float zn = zo + DT * (-adv - BETA * v - R_DRAG * zo + Q[idx]);
  z[idx] = zn;
  if (oidx >= 0)
    out[((b * 4 + oidx) * 256 + y) * 256 + t] = zn;
}

// ---------------------------------------------------------------------------
// Host launcher: K0 ; then 16 steps of [K12 -> K3 -> K4] (4 outputs x 4 steps).
// ws layout (floats): z[512K] | Q[512K] | Zt[1M] | S1[1M] | S2[1M]  (16 MB)
// ---------------------------------------------------------------------------
extern "C" void kernel_launch(void* const* d_in, const int* in_sizes, int n_in,
                              void* d_out, int out_size, void* d_ws, size_t ws_size,
                              hipStream_t stream)
{
  (void)in_sizes; (void)n_in; (void)out_size; (void)ws_size;
  const float* zeta = (const float*)d_in[0];
  const float* Q0   = (const float*)d_in[1];
  const float* W1   = (const float*)d_in[2];
  const float* b1   = (const float*)d_in[3];
  const float* W2   = (const float*)d_in[4];
  const float* b2   = (const float*)d_in[5];
  float* outp = (float*)d_out;

  float*  ws   = (float*)d_ws;
  float*  zbuf = ws;                               // BATCH*NPIX
  float*  Qbuf = zbuf + BATCH * NPIX;              // BATCH*NPIX
  float2* Ztb  = (float2*)(Qbuf + BATCH * NPIX);   // BATCH*NPIX float2 (transposed)
  float2* S1b  = Ztb + BATCH * NPIX;
  float2* S2b  = S1b + BATCH * NPIX;

  k0_init<<<dim3(BATCH * NPIX / 256), dim3(256), 0, stream>>>(zeta, zbuf);

  for (int o = 0; o < 4; ++o) {
    for (int s = 0; s < 4; ++s) {
      k12_cnn_rows<<<dim3(272, BATCH), dim3(256), 0, stream>>>(
          zbuf, Q0, W1, b1, W2, b2, Qbuf, Ztb);
      k3_cols<<<dim3(32, BATCH), dim3(256), 0, stream>>>(Ztb, S1b, S2b);
      k4_rows_inv_update<<<dim3(GRIDN, BATCH), dim3(256), 0, stream>>>(
          S1b, S2b, Qbuf, zbuf, outp, (s == 3) ? o : -1);
    }
  }
}
